// LlamaAttentionWithLora_88313117541048
// MI455X (gfx1250) — matrix operand order repaired and checked by reading, not compiled, not run on hardware
//
#include <hip/hip_runtime.h>
#include <hip/hip_bf16.h>

// ---------------------------------------------------------------------------
// Problem constants (match reference)
// ---------------------------------------------------------------------------
#define BB   4
#define SS   1024
#define HH   32
#define DD   128
#define HID  4096
#define TT   (BB * SS)       // 4096 tokens
#define LL   8
#define RR   16
#define ATT_SCALE 0.08838834764831845f   // 1/sqrt(128)
#define LORA_SCALE_C 1.0f

typedef __bf16 bf16_t;
typedef __attribute__((ext_vector_type(16))) __bf16 v16bf;
typedef __attribute__((ext_vector_type(8)))  float  v8f;

union ABFrag { v16bf v; uint4 u[2]; };

__device__ inline bf16_t f2bf(float f) {
  union { float f; unsigned u; } in; in.f = f;
  unsigned u = in.u;
  unsigned r = (u + 0x7FFFu + ((u >> 16) & 1u)) >> 16;   // round-nearest-even
  union { unsigned short s; bf16_t b; } o; o.s = (unsigned short)r;
  return o.b;
}
__device__ inline float b2f(bf16_t b) {
  union { unsigned short s; bf16_t b; } i; i.b = b;
  union { unsigned u; float f; } o; o.u = ((unsigned)i.s) << 16;
  return o.f;
}
__device__ inline float tofloat(float f)  { return f; }
__device__ inline float tofloat(bf16_t b) { return b2f(b); }

__device__ inline v8f wmma_bf16(v16bf a, v16bf b, v8f c) {
  return __builtin_amdgcn_wmma_f32_16x16x32_bf16(false, a, false, b,
                                                 (short)0, c, false, false);
}

// ---- CDNA5 TDM / async staging support (guarded, cannot break build) ------
#if defined(__has_builtin)
#if __has_builtin(__builtin_amdgcn_tensor_load_to_lds) && \
    __has_builtin(__builtin_amdgcn_s_wait_tensorcnt)
#define HAVE_TDM 1
#endif
#if __has_builtin(__builtin_amdgcn_load_to_lds)
#define HAVE_LOAD_TO_LDS 1
#endif
#if __has_builtin(__builtin_amdgcn_s_wait_asynccnt)
#define HAVE_WAIT_ASYNC_BUILTIN 1
#endif
#endif

__device__ inline void wait_async0() {
#if defined(HAVE_WAIT_ASYNC_BUILTIN)
  __builtin_amdgcn_s_wait_asynccnt(0);
#else
  asm volatile("s_wait_asynccnt 0x0" ::: "memory");
#endif
}

#if defined(HAVE_TDM)
typedef __attribute__((ext_vector_type(4))) unsigned tdm_v4u;
typedef __attribute__((ext_vector_type(8))) int      tdm_v8i;
typedef __attribute__((ext_vector_type(4))) int      tdm_v4i;

__device__ inline unsigned lds_offset_of(const void* p) {
  return (unsigned)(uintptr_t)(__attribute__((address_space(3))) const void*)p;
}

// DMA one 128-row x 64-elem bf16 tile (row stride = K elems) into LDS with
// hardware row padding: pad 4 DWORDs (16B) after every 32 DWORDs (128B row)
// -> matches the 72-element G_PITCH layout used by the WMMA fragment loads.
__device__ inline void tdm_load_tile(const bf16_t* gsrc, unsigned lds_off, int K) {
  unsigned long long ga = (unsigned long long)(uintptr_t)gsrc;
  tdm_v4u g0;
  g0[0] = 1u;                                        // count=1 (valid user D#)
  g0[1] = lds_off;                                   // lds_addr (bytes)
  g0[2] = (unsigned)(ga & 0xFFFFFFFFu);              // global_addr[31:0]
  g0[3] = (unsigned)((ga >> 32) & 0x01FFFFFFu)       // global_addr[56:32]
          | (2u << 30);                              // type=2 ("image")
  tdm_v8i g1;
  g1[0] = (int)((1u << 16)      // data_size = 1 -> 2 bytes
              | (1u << 20)      // pad_enable
              | (4u << 22)      // pad_interval: 4 -> 32 DWORDs
              | (3u << 25));    // pad_amount:   3 -> 4 DWORDs
  g1[1] = (int)(((unsigned)K & 0xFFFFu) << 16);      // tensor_dim0[15:0]
  g1[2] = (int)((((unsigned)K >> 16) & 0xFFFFu)      // tensor_dim0[31:16]
              | (((unsigned)K & 0xFFFFu) << 16));    // tensor_dim1[15:0] (=K rows avail)
  g1[3] = (int)((((unsigned)K >> 16) & 0xFFFFu)      // tensor_dim1[31:16]
              | (64u << 16));                        // tile_dim0 = 64 elems
  g1[4] = (int)(128u);                               // tile_dim1 = 128 rows, tile_dim2=0
  g1[5] = (int)(unsigned)K;                          // tensor_dim0_stride[31:0]
  g1[6] = 0;                                         // stride hi / dim1_stride lo
  g1[7] = 0;
  tdm_v4i gz4 = (tdm_v4i){0, 0, 0, 0};
  tdm_v8i gz8 = (tdm_v8i){0, 0, 0, 0, 0, 0, 0, 0};
  // 6-arg form (amdgpu-toolchain clang-23): (g0, g1, g2, g3, g4, cpol)
  __builtin_amdgcn_tensor_load_to_lds(g0, g1, gz4, gz4, gz8, 0);
}
#endif

// ---------------------------------------------------------------------------
// fp32 -> bf16 convert (4-wide)
// ---------------------------------------------------------------------------
__global__ void cvt_f32_bf16_kernel(const float* __restrict__ src,
                                    bf16_t* __restrict__ dst, int n4) {
  int i = blockIdx.x * blockDim.x + threadIdx.x;
  if (i >= n4) return;
  int idx = i * 4;
  float4 v = *(const float4*)(src + idx);
  bf16_t o[4] = { f2bf(v.x), f2bf(v.y), f2bf(v.z), f2bf(v.w) };
  *(uint2*)(dst + idx) = *(const uint2*)o;
}

// ---------------------------------------------------------------------------
// GEMM: C[M,N] = A[M,K] * W[N,K]^T   (A,W bf16 row-major, C fp32)
// block 256 thr = 8 waves; block tile 128x128; wave tile 32x64; BK=64
// Double-buffered LDS; TDM (tensor_load_to_lds) staging when available.
// ---------------------------------------------------------------------------
#define G_BM 128
#define G_BN 128
#define G_BK 64
#define G_PITCH 72   // bf16 elems per LDS row (64 + 8 pad, keeps 16B align)

__global__ __launch_bounds__(256)
void gemm_bf16_nt_kernel(const bf16_t* __restrict__ A,
                         const bf16_t* __restrict__ W,
                         float* __restrict__ C,
                         int M, int N, int K) {
  __shared__ bf16_t sA[2][G_BM * G_PITCH];
  __shared__ bf16_t sW[2][G_BN * G_PITCH];

  const int tid  = threadIdx.x;
  const int lane = tid & 31;
  const int wave = tid >> 5;
  const int hi   = lane >> 4;
  const int ln   = lane & 15;
  const int wm   = wave & 3;     // 4 waves along M
  const int wn   = wave >> 2;    // 2 waves along N
  const int m0   = blockIdx.y * G_BM;
  const int n0   = blockIdx.x * G_BN;

  v8f acc[2][4];
#pragma unroll
  for (int mt = 0; mt < 2; ++mt)
#pragma unroll
    for (int nt = 0; nt < 4; ++nt)
      acc[mt][nt] = (v8f){0.f,0.f,0.f,0.f,0.f,0.f,0.f,0.f};

  const int lrow = tid >> 1;          // 0..127
  const int lcol = (tid & 1) * 32;    // 0 or 32 bf16 elems

  auto compute = [&](int buf) {
#pragma unroll
    for (int kk = 0; kk < G_BK; kk += 32) {
      v16bf fa[2], fb[4];
#pragma unroll
      for (int mt = 0; mt < 2; ++mt) {
        const bf16_t* base = &sA[buf][(wm * 32 + mt * 16 + ln) * G_PITCH];
        ABFrag f;
        f.u[0] = *(const uint4*)(base + kk + 8 * hi);
        f.u[1] = *(const uint4*)(base + kk + 16 + 8 * hi);
        fa[mt] = f.v;
      }
#pragma unroll
      for (int nt = 0; nt < 4; ++nt) {
        const bf16_t* base = &sW[buf][(wn * 64 + nt * 16 + ln) * G_PITCH];
        ABFrag f;
        f.u[0] = *(const uint4*)(base + kk + 8 * hi);
        f.u[1] = *(const uint4*)(base + kk + 16 + 8 * hi);
        fb[nt] = f.v;
      }
#pragma unroll
      for (int mt = 0; mt < 2; ++mt)
#pragma unroll
        for (int nt = 0; nt < 4; ++nt)
          acc[mt][nt] = wmma_bf16(fa[mt], fb[nt], acc[mt][nt]);
    }
  };

#if defined(HAVE_TDM)
  // --- Tensor Data Mover double-buffered pipeline -------------------------
  // wave 0 issues two tile DMAs (A,W) per K-step; TENSORcnt tracks them.
  const unsigned ldsA0 = lds_offset_of(&sA[0][0]);
  const unsigned ldsA1 = lds_offset_of(&sA[1][0]);
  const unsigned ldsW0 = lds_offset_of(&sW[0][0]);
  const unsigned ldsW1 = lds_offset_of(&sW[1][0]);
  if (wave == 0) {
    tdm_load_tile(A + (size_t)m0 * K, ldsA0, K);
    tdm_load_tile(W + (size_t)n0 * K, ldsW0, K);
  }
  int buf = 0;
  for (int k0 = 0; k0 < K; k0 += G_BK) {
    __syncthreads();                 // everyone done reading buf^1
    if (wave == 0) {
      if (k0 + G_BK < K) {           // kick next tile, then wait oldest tile
        tdm_load_tile(A + (size_t)m0 * K + k0 + G_BK, buf ? ldsA0 : ldsA1, K);
        tdm_load_tile(W + (size_t)n0 * K + k0 + G_BK, buf ? ldsW0 : ldsW1, K);
        __builtin_amdgcn_s_wait_tensorcnt(2);
      } else {
        __builtin_amdgcn_s_wait_tensorcnt(0);
      }
    }
    __syncthreads();                 // publish tile `buf` to all waves
    compute(buf);
    buf ^= 1;
  }
#else
  // --- fallback: staged through VGPRs -------------------------------------
  auto stage = [&](int buf, int k0) {
    const bf16_t* ga = A + (size_t)(m0 + lrow) * K + k0 + lcol;
    const bf16_t* gw = W + (size_t)(n0 + lrow) * K + k0 + lcol;
    bf16_t* la = &sA[buf][lrow * G_PITCH + lcol];
    bf16_t* lw = &sW[buf][lrow * G_PITCH + lcol];
#if defined(HAVE_LOAD_TO_LDS)
#pragma unroll
    for (int i = 0; i < 4; ++i) {
      __builtin_amdgcn_load_to_lds(
          (void*)(ga + i * 8),
          (__attribute__((address_space(3))) void*)(la + i * 8), 16, 0, 0);
      __builtin_amdgcn_load_to_lds(
          (void*)(gw + i * 8),
          (__attribute__((address_space(3))) void*)(lw + i * 8), 16, 0, 0);
    }
#else
    uint4 ra[4], rw[4];
#pragma unroll
    for (int i = 0; i < 4; ++i) {
      ra[i] = *(const uint4*)(ga + i * 8);
      rw[i] = *(const uint4*)(gw + i * 8);
    }
#pragma unroll
    for (int i = 0; i < 4; ++i) {
      *(uint4*)(la + i * 8) = ra[i];
      *(uint4*)(lw + i * 8) = rw[i];
    }
#endif
  };
  for (int k0 = 0; k0 < K; k0 += G_BK) {
    __syncthreads();
    stage(0, k0);
#if defined(HAVE_LOAD_TO_LDS)
    wait_async0();
#endif
    __syncthreads();
    if (k0 + G_BK < K) {    // prefetch next tile toward L2
      __builtin_prefetch(A + (size_t)(m0 + lrow) * K + k0 + G_BK + lcol, 0, 3);
      __builtin_prefetch(W + (size_t)(n0 + lrow) * K + k0 + G_BK + lcol, 0, 3);
    }
    compute(0);
  }
#endif

#pragma unroll
  for (int mt = 0; mt < 2; ++mt)
#pragma unroll
    for (int nt = 0; nt < 4; ++nt)
#pragma unroll
      for (int r = 0; r < 8; ++r) {
        int row = m0 + wm * 32 + mt * 16 + hi * 8 + r;
        int col = n0 + wn * 64 + nt * 16 + ln;
        C[(size_t)row * N + col] = acc[mt][nt][r];
      }
}

// ---------------------------------------------------------------------------
// LoRA A: a[t,r] = sum_h X[t,h] * wa[idx[t], r, h]     block=(32,16), grid=T
// ---------------------------------------------------------------------------
template <typename T>
__global__ void lora_a_kernel(const T* __restrict__ X,
                              const float* __restrict__ wa,
                              const int* __restrict__ idx,
                              float* __restrict__ out) {
  const int t = blockIdx.x;
  const int r = threadIdx.y;
  const int lane = threadIdx.x;
  const int l = idx[t];
  const float* w = wa + ((size_t)l * RR + r) * HID;
  const T* x = X + (size_t)t * HID;
  float s = 0.f;
  for (int h = lane; h < HID; h += 32) s += tofloat(x[h]) * w[h];
#pragma unroll
  for (int m = 16; m; m >>= 1) s += __shfl_xor(s, m, 32);
  if (lane == 0) out[t * RR + r] = s;
}

// ---------------------------------------------------------------------------
// LoRA B add: Y[t,o] += scale * sum_r a[t,r] * wb[idx[t], o, r]
// block=256, grid=(HID/256, T)
// ---------------------------------------------------------------------------
__global__ void lora_b_add_kernel(const float* __restrict__ a,
                                  const float* __restrict__ wb,
                                  const int* __restrict__ idx,
                                  float* __restrict__ Y, float scale) {
  const int t = blockIdx.y;
  const int o = blockIdx.x * 256 + threadIdx.x;
  const int l = idx[t];
  const float4* w4 = (const float4*)(wb + ((size_t)l * HID + o) * RR);
  const float4* a4 = (const float4*)(a + (size_t)t * RR);
  float s = 0.f;
#pragma unroll
  for (int i = 0; i < 4; ++i) {
    float4 w = w4[i], av = a4[i];
    s += w.x * av.x + w.y * av.y + w.z * av.z + w.w * av.w;
  }
  Y[(size_t)t * HID + o] += scale * s;
}

// ---------------------------------------------------------------------------
// RoPE + layout: src f32 [B,S,H,D] -> dst bf16 [B,H,S,D]
// one thread per (t,h,i<64): handles the rotate-half pair (i, i+64)
// ---------------------------------------------------------------------------
__global__ void rope_cvt_kernel(const float* __restrict__ src,
                                bf16_t* __restrict__ dst) {
  int gid = blockIdx.x * blockDim.x + threadIdx.x;    // over T*H*64
  int i = gid & 63;
  int h = (gid >> 6) & (HH - 1);
  int t = gid >> 11;
  int s = t & (SS - 1);
  int b = t >> 10;
  float x1 = src[(size_t)t * HID + h * DD + i];
  float x2 = src[(size_t)t * HID + h * DD + i + 64];
  float invf = __powf(10000.f, -(float)i * (1.f / 64.f));
  float ang = (float)s * invf;
  float sn, c;
  __sincosf(ang, &sn, &c);
  size_t d0 = (((size_t)b * HH + h) * SS + s) * DD + i;
  dst[d0]      = f2bf(x1 * c - x2 * sn);
  dst[d0 + 64] = f2bf(x2 * c + x1 * sn);
}

// ---------------------------------------------------------------------------
// V convert + transpose: src f32 [B,S,H,D] -> dst bf16 [B,H,D,S]
// ---------------------------------------------------------------------------
__global__ void v_cvt_kernel(const float* __restrict__ src,
                             bf16_t* __restrict__ dst) {
  int gid = blockIdx.x * blockDim.x + threadIdx.x;    // over T*HID
  int d = gid & (DD - 1);
  int h = (gid >> 7) & (HH - 1);
  int t = gid >> 12;
  int s = t & (SS - 1);
  int b = t >> 10;
  dst[(((size_t)b * HH + h) * DD + d) * SS + s] = f2bf(src[gid]);
}

// ---------------------------------------------------------------------------
// Causal attention. One wave = one 16-query tile of one (b,h).
// Q,K: bf16 [B,H,S,D]; Vt: bf16 [B,H,D,S]; O: bf16 [T, HID] (== [B,S,H,D])
// block = 128 thr (4 waves); dynamic LDS = 4 * 16*1024 fp32 score strips.
// ---------------------------------------------------------------------------
__global__ __launch_bounds__(128)
void attn_kernel(const bf16_t* __restrict__ Q, const bf16_t* __restrict__ Kc,
                 const bf16_t* __restrict__ Vt, bf16_t* __restrict__ O) {
  extern __shared__ char smem_raw[];
  const int lane = threadIdx.x & 31;
  const int wave = threadIdx.x >> 5;
  const int hi = lane >> 4, ln = lane & 15;
  const int bh = blockIdx.x;                 // 0..B*H-1
  const int b = bh >> 5, h = bh & (HH - 1);
  const int qt = blockIdx.y * 4 + wave;      // q tile 0..63
  const int q0 = qt * 16;

  float* sc = (float*)smem_raw + (size_t)wave * 16 * SS;

  const bf16_t* Qb = Q + (size_t)bh * SS * DD;
  const bf16_t* Kb = Kc + (size_t)bh * SS * DD;
  const bf16_t* Vb = Vt + (size_t)bh * DD * SS;

  // Q fragments, kept in registers (4 K-substeps over D=128)
  v16bf fq[4];
  {
    const bf16_t* qrow = Qb + (size_t)(q0 + ln) * DD;
#pragma unroll
    for (int c = 0; c < 4; ++c) {
      ABFrag f;
      f.u[0] = *(const uint4*)(qrow + c * 32 + 8 * hi);
      f.u[1] = *(const uint4*)(qrow + c * 32 + 16 + 8 * hi);
      fq[c] = f.v;
    }
  }

  // --- scores: Q K^T, causal-limited
  const int nkb = qt + 1;
  for (int kb = 0; kb < nkb; ++kb) {
    v8f acc = (v8f){0.f,0.f,0.f,0.f,0.f,0.f,0.f,0.f};
    const bf16_t* krow = Kb + (size_t)(kb * 16 + ln) * DD;
#pragma unroll
    for (int c = 0; c < 4; ++c) {
      ABFrag f;
      f.u[0] = *(const uint4*)(krow + c * 32 + 8 * hi);
      f.u[1] = *(const uint4*)(krow + c * 32 + 16 + 8 * hi);
      acc = wmma_bf16(fq[c], f.v, acc);
    }
    int kg = kb * 16 + ln;
#pragma unroll
    for (int r = 0; r < 8; ++r) {
      int m = hi * 8 + r;
      float v = acc[r] * ATT_SCALE;
      if (kg > q0 + m) v = -1e30f;
      sc[m * SS + kg] = v;
    }
  }
  const int kmax = q0 + 16;
  const int ncol = (kmax + 31) & ~31;
  if (ncol != kmax) {          // pad diag tile to 32-wide for the PV pass
    for (int e = lane; e < 16 * 16; e += 32) {
      int m = e >> 4, c = e & 15;
      sc[m * SS + kmax + c] = -1e30f;
    }
  }
  __syncthreads();

  // --- softmax, row by row, in place
  for (int m = 0; m < 16; ++m) {
    float* row = sc + m * SS;
    float mx = -1e30f;
    for (int c = lane; c < ncol; c += 32) mx = fmaxf(mx, row[c]);
#pragma unroll
    for (int t = 16; t; t >>= 1) mx = fmaxf(mx, __shfl_xor(mx, t, 32));
    float sm = 0.f;
    for (int c = lane; c < ncol; c += 32) sm += __expf(row[c] - mx);
#pragma unroll
    for (int t = 16; t; t >>= 1) sm += __shfl_xor(sm, t, 32);
    float inv = 1.f / sm;
    for (int c = lane; c < ncol; c += 32) row[c] = __expf(row[c] - mx) * inv;
  }
  __syncthreads();

  // --- P V : accumulate 16x128 output
  v8f acc[8];
#pragma unroll
  for (int dt = 0; dt < 8; ++dt) acc[dt] = (v8f){0.f,0.f,0.f,0.f,0.f,0.f,0.f,0.f};

  for (int k0 = 0; k0 < ncol; k0 += 32) {
    v16bf fa;
    const float* prow = sc + ln * SS + k0;   // row m = ln
#pragma unroll
    for (int j = 0; j < 8; ++j) {
      fa[j]     = f2bf(prow[16 * hi + j]);
      fa[8 + j] = f2bf(prow[16 * hi + 8 + j]);
    }
#pragma unroll
    for (int dt = 0; dt < 8; ++dt) {
      const bf16_t* vrow = Vb + (size_t)(dt * 16 + ln) * SS + k0 + 16 * hi;
      ABFrag f;
      f.u[0] = ((const uint4*)vrow)[0];
      f.u[1] = ((const uint4*)vrow)[1];
      acc[dt] = wmma_bf16(fa, f.v, acc[dt]);
    }
  }

#pragma unroll
  for (int dt = 0; dt < 8; ++dt)
#pragma unroll
    for (int r = 0; r < 8; ++r) {
      int m = hi * 8 + r;
      size_t t = (size_t)b * SS + q0 + m;
      O[t * HID + h * DD + dt * 16 + ln] = f2bf(acc[dt][r]);
    }
}

// ---------------------------------------------------------------------------
// Host-side orchestration
// ---------------------------------------------------------------------------
extern "C" void kernel_launch(void* const* d_in, const int* in_sizes, int n_in,
                              void* d_out, int out_size, void* d_ws, size_t ws_size,
                              hipStream_t stream) {
  (void)in_sizes; (void)n_in; (void)out_size; (void)ws_size;
  const float* x  = (const float*)d_in[0];
  const int* idx  = (const int*)d_in[1];
  const float* Wq = (const float*)d_in[2];
  const float* Wk = (const float*)d_in[3];
  const float* Wv = (const float*)d_in[4];
  const float* Wo = (const float*)d_in[5];
  const float* wa_q = (const float*)d_in[6];  const float* wb_q = (const float*)d_in[7];
  const float* wa_k = (const float*)d_in[8];  const float* wb_k = (const float*)d_in[9];
  const float* wa_v = (const float*)d_in[10]; const float* wb_v = (const float*)d_in[11];
  const float* wa_o = (const float*)d_in[12]; const float* wb_o = (const float*)d_in[13];
  float* out = (float*)d_out;

  char* ws = (char*)d_ws;
  size_t off = 0;
  auto alloc = [&](size_t bytes) -> void* {
    void* p = ws + off;
    off = (off + bytes + 255) & ~(size_t)255;
    return p;
  };
  const size_t NE = (size_t)TT * HID;          // 16.7M elements
  bf16_t* xb     = (bf16_t*)alloc(NE * 2);     // x bf16
  bf16_t* wbw    = (bf16_t*)alloc(NE * 2);     // current weight bf16 (reused 4x)
  float*  tmpf   = (float*)alloc(NE * 4);      // q/k/v fp32 (reused 3x)
  bf16_t* q_b    = (bf16_t*)alloc(NE * 2);     // [B,H,S,D]
  bf16_t* k_b    = (bf16_t*)alloc(NE * 2);     // [B,H,S,D]
  bf16_t* vt_b   = (bf16_t*)alloc(NE * 2);     // [B,H,D,S]
  bf16_t* attn_b = (bf16_t*)alloc(NE * 2);     // [T, HID]
  float*  a_sel  = (float*)alloc((size_t)TT * RR * 4);

  const int n4 = (int)(NE / 4);
  const dim3 cvt_grid((n4 + 255) / 256);
  const dim3 gemm_grid(HID / G_BN, TT / G_BM);
  const dim3 la_block(32, 16);
  const dim3 lb_grid(HID / 256, TT);

  // x -> bf16
  cvt_f32_bf16_kernel<<<cvt_grid, 256, 0, stream>>>(x, xb, n4);

  // ---- Q ----
  cvt_f32_bf16_kernel<<<cvt_grid, 256, 0, stream>>>(Wq, wbw, n4);
  gemm_bf16_nt_kernel<<<gemm_grid, 256, 0, stream>>>(xb, wbw, tmpf, TT, HID, HID);
  lora_a_kernel<float><<<TT, la_block, 0, stream>>>(x, wa_q, idx, a_sel);
  lora_b_add_kernel<<<lb_grid, 256, 0, stream>>>(a_sel, wb_q, idx, tmpf, LORA_SCALE_C);
  rope_cvt_kernel<<<(TT * HH * 64) / 256, 256, 0, stream>>>(tmpf, q_b);

  // ---- K ----
  cvt_f32_bf16_kernel<<<cvt_grid, 256, 0, stream>>>(Wk, wbw, n4);
  gemm_bf16_nt_kernel<<<gemm_grid, 256, 0, stream>>>(xb, wbw, tmpf, TT, HID, HID);
  lora_a_kernel<float><<<TT, la_block, 0, stream>>>(x, wa_k, idx, a_sel);
  lora_b_add_kernel<<<lb_grid, 256, 0, stream>>>(a_sel, wb_k, idx, tmpf, LORA_SCALE_C);
  rope_cvt_kernel<<<(TT * HH * 64) / 256, 256, 0, stream>>>(tmpf, k_b);

  // ---- V ----
  cvt_f32_bf16_kernel<<<cvt_grid, 256, 0, stream>>>(Wv, wbw, n4);
  gemm_bf16_nt_kernel<<<gemm_grid, 256, 0, stream>>>(xb, wbw, tmpf, TT, HID, HID);
  lora_a_kernel<float><<<TT, la_block, 0, stream>>>(x, wa_v, idx, a_sel);
  lora_b_add_kernel<<<lb_grid, 256, 0, stream>>>(a_sel, wb_v, idx, tmpf, LORA_SCALE_C);
  v_cvt_kernel<<<(int)(NE / 256), 256, 0, stream>>>(tmpf, vt_b);

  // ---- attention ----
  attn_kernel<<<dim3(BB * HH, SS / 16 / 4), 128, 4 * 16 * SS * sizeof(float), stream>>>(
      q_b, k_b, vt_b, attn_b);

  // ---- output projection ----
  cvt_f32_bf16_kernel<<<cvt_grid, 256, 0, stream>>>(Wo, wbw, n4);
  gemm_bf16_nt_kernel<<<gemm_grid, 256, 0, stream>>>(attn_b, wbw, out, TT, HID, HID);
  lora_a_kernel<bf16_t><<<TT, la_block, 0, stream>>>(attn_b, wa_o, idx, a_sel);
  lora_b_add_kernel<<<lb_grid, 256, 0, stream>>>(a_sel, wb_o, idx, out, LORA_SCALE_C);
}
